// RNNSimple_17566416241292
// MI455X (gfx1250) — matrix operand order repaired and checked by reading, compiled, not run on hardware
//
#include <hip/hip_runtime.h>

// ---------------------------------------------------------------------------
// RNN fused kernel for MI455X (gfx1250), round 2.
//  - bf16 WMMA (v_wmma_f32_16x16x32_bf16), fp32 accumulate.
//  - batch-partitioned: each WG owns 32 batch columns; h tile ([1024x32] bf16)
//    ping-pongs in LDS across T=32 steps (no global h round trips).
//  - Double-buffered W k-slab, staged via registers: global-load chunk kc+1,
//    WMMA-compute chunk kc, ds-store chunk kc+1, ONE barrier per chunk.
//  - 1024 threads (32 wave32 waves): 4 m-tiles/wave -> low register pressure,
//    lets the compiler pipeline ds_loads against WMMAs.
// ---------------------------------------------------------------------------

#define IN_DIM   512
#define HID      1024
#define OUT_DIM  512
#define BATCH    4096
#define TSTEPS   32

#define BT       32      // batch columns per workgroup
#define NTHREADS 1024    // 32 wave32 waves
#define HSTRIDE  1032    // h tile row stride in halfs (16B aligned, bank-spread)
#define WSTRIDE  40      // W slab row stride in halfs (80B -> 16B aligned)

typedef __attribute__((ext_vector_type(16))) __bf16 v16bf;
typedef __attribute__((ext_vector_type(8)))  __bf16 v8bf;
typedef __attribute__((ext_vector_type(8)))  float  v8f;

__device__ __forceinline__ v16bf make16(v8bf lo, v8bf hi) {
  v16bf r;
#pragma unroll
  for (int e = 0; e < 8; ++e) { r[e] = lo[e]; r[8 + e] = hi[e]; }
  return r;
}

// A fragment (16x32 bf16, MxK): lane holds row m = m0 + (lane&15);
// lanes 0-15 carry K 0..7 & 16..23, lanes 16-31 carry K 8..15 & 24..31.
__device__ __forceinline__ v16bf load_A(const __bf16* wslab, int m0, int lane) {
  const __bf16* row = wslab + (m0 + (lane & 15)) * WSTRIDE;
  const int h8 = (lane >> 4) * 8;
  v8bf lo = *(const v8bf*)(row + h8);
  v8bf hi = *(const v8bf*)(row + 16 + h8);
  return make16(lo, hi);
}

// B fragment (32x16 bf16, KxN): lane holds col n; lanes 0-15 carry
// K = kc*32+0..7 & 16..23, lanes 16-31 carry K = kc*32+8..15 & 24..31. h is [n][k] in LDS.
__device__ __forceinline__ v16bf load_B(const __bf16* hrow, int kc, int hf) {
  const __bf16* p = hrow + kc * 32 + hf * 8;
  v8bf lo = *(const v8bf*)(p);
  v8bf hi = *(const v8bf*)(p + 16);
  return make16(lo, hi);
}

// One GEMM phase: D = relu(W @ h + bias).  MTILES = 16x16 output tiles per
// wave.  W streamed through double-buffered LDS slabs with a register-staged
// software pipeline; h read from LDS [n][k]; results written to global fp32
// and optionally back to LDS as bf16 for the next step.
template <int MTILES>
__device__ __forceinline__ void gemm_phase(
    const __bf16* __restrict__ Wg, int Kdim, int Mrows,
    const __bf16* __restrict__ hsrc, __bf16* wsA, __bf16* wsB,
    const float* __restrict__ bias, float* __restrict__ gdst,
    __bf16* hdst, int tid, int lane, int col0) {
  const int hf     = lane >> 4;
  const int ln15   = lane & 15;
  const int wave   = tid >> 5;
  const int n0     = (wave & 1) * 16;
  const int mslice = wave >> 1;                 // 0..15
  const int mwave  = mslice * (MTILES * 16);    // wave's first output row
  const int nkc    = Kdim / 32;
  const bool stager = tid < Mrows;              // one W row per thread

  v8f acc[MTILES];
#pragma unroll
  for (int i = 0; i < MTILES; ++i)
#pragma unroll
    for (int r = 0; r < 8; ++r) acc[i][r] = 0.0f;

  const __bf16* hrow = hsrc + (n0 + ln15) * HSTRIDE;
  const uint4* wsrc = (const uint4*)(Wg + (size_t)tid * Kdim);

  // prologue: stage chunk 0 into wsA
  if (stager) {
    uint4 r0 = wsrc[0], r1 = wsrc[1], r2 = wsrc[2], r3 = wsrc[3];
    uint4* dst = (uint4*)(wsA + tid * WSTRIDE);
    dst[0] = r0; dst[1] = r1; dst[2] = r2; dst[3] = r3;
  }
  __syncthreads();

  for (int kc = 0; kc < nkc; ++kc) {
    __bf16* slab = (kc & 1) ? wsB : wsA;
    __bf16* next = (kc & 1) ? wsA : wsB;

    // issue global loads for chunk kc+1 (latency hidden behind WMMAs)
    uint4 r0, r1, r2, r3;
    const bool ld = stager && (kc + 1 < nkc);
    if (ld) {
      const uint4* src = wsrc + (size_t)(kc + 1) * 4;  // +32 halfs
      r0 = src[0]; r1 = src[1]; r2 = src[2]; r3 = src[3];
      __builtin_prefetch(wsrc + (size_t)(kc + 2) * 4, 0, 3);
    }

    // compute chunk kc: 1 B fragment, MTILES A fragments (pipelined)
    v16bf bfr = load_B(hrow, kc, hf);
    v16bf a0 = load_A(slab, mwave, lane);
#pragma unroll
    for (int i = 0; i < MTILES; ++i) {
      v16bf an = a0;
      if (i + 1 < MTILES) an = load_A(slab, mwave + (i + 1) * 16, lane);
      acc[i] = __builtin_amdgcn_wmma_f32_16x16x32_bf16(
          false, a0, false, bfr, (short)0, acc[i], false, false);
      a0 = an;
    }

    // commit chunk kc+1 into the other slab, then one barrier
    if (ld) {
      uint4* dst = (uint4*)(next + tid * WSTRIDE);
      dst[0] = r0; dst[1] = r1; dst[2] = r2; dst[3] = r3;
    }
    __syncthreads();
  }

  // epilogue: bias + relu -> global fp32, and bf16 h for the next step
#pragma unroll
  for (int i = 0; i < MTILES; ++i) {
    const int mbase = mwave + i * 16;
#pragma unroll
    for (int r = 0; r < 8; ++r) {
      const int m = mbase + r + 8 * hf;
      float v = acc[i][r] + bias[m];
      v = v > 0.0f ? v : 0.0f;
      gdst[(size_t)m * BATCH + col0 + n0 + ln15] = v;
      if (hdst) hdst[(n0 + ln15) * HSTRIDE + m] = (__bf16)v;
    }
  }
}

__global__ __launch_bounds__(NTHREADS, 1)
void rnn_fused(const float*  __restrict__ x,      // [512, 4096]
               const __bf16* __restrict__ Winb,   // [1024, 512]
               const float*  __restrict__ b_in,   // [1024]
               const __bf16* __restrict__ Wrecb,  // [1024, 1024]
               const float*  __restrict__ b_rec,  // [1024]
               const __bf16* __restrict__ Woutb,  // [512, 1024]
               const float*  __restrict__ b_out,  // [512]
               float* __restrict__ out,           // [512, 4096]
               float* __restrict__ interm) {      // [33, 1024, 4096]
  extern __shared__ __align__(16) char smem[];
  __bf16* hA  = (__bf16*)smem;              // 32 x 1032 halfs (64.5 KB)
  __bf16* hB  = hA + BT * HSTRIDE;          // 32 x 1032 halfs
  __bf16* wsA = hB + BT * HSTRIDE;          // 1024 x 40 halfs (80 KB)
  __bf16* wsB = wsA + HID * WSTRIDE;        // 1024 x 40 halfs

  const int tid  = threadIdx.x;
  const int lane = tid & 31;
  const int col0 = blockIdx.x * BT;

  // stage x tile (fp32 global, k-major over batch) as bf16 [n][k] into hB
  {
    const int n = tid & 31, ks = tid >> 5;  // 32 k-slices
#pragma unroll
    for (int j = 0; j < IN_DIM / 32; ++j) {
      const int k = ks + 32 * j;
      hB[n * HSTRIDE + k] = (__bf16)x[(size_t)k * BATCH + col0 + n];
    }
  }
  __syncthreads();

  // Phase A: y0 = relu(W_in @ x + b_in)  -> interm[0], hA
  gemm_phase<4>(Winb, IN_DIM, HID, hB, wsA, wsB, b_in, interm, hA,
                tid, lane, col0);

  // Phase B: T recurrent steps, h ping-pongs between hA/hB
  const __bf16* hcur = hA;
  __bf16*       hnxt = hB;
  for (int t = 1; t <= TSTEPS; ++t) {
    gemm_phase<4>(Wrecb, HID, HID, hcur, wsA, wsB, b_rec,
                  interm + (size_t)t * HID * BATCH, hnxt, tid, lane, col0);
    const __bf16* tmp = hcur; hcur = hnxt; hnxt = (__bf16*)tmp;
  }

  // Phase C: out = relu(W_out @ h_last + b_out)
  gemm_phase<2>(Woutb, HID, OUT_DIM, hcur, wsA, wsB, b_out, out, nullptr,
                tid, lane, col0);
}

// fp32 -> bf16 one-shot weight conversion (memory-bound, ~8MB total)
__global__ void f2bf(const float* __restrict__ src, __bf16* __restrict__ dst,
                     int n) {
  int i = blockIdx.x * blockDim.x + threadIdx.x;
  const int stride = gridDim.x * blockDim.x;
  for (; i < n; i += stride) dst[i] = (__bf16)src[i];
}

extern "C" void kernel_launch(void* const* d_in, const int* in_sizes, int n_in,
                              void* d_out, int out_size, void* d_ws,
                              size_t ws_size, hipStream_t stream) {
  const float* x     = (const float*)d_in[0];
  const float* W_in  = (const float*)d_in[1];
  const float* b_in  = (const float*)d_in[2];
  const float* W_rec = (const float*)d_in[3];
  const float* b_rec = (const float*)d_in[4];
  const float* W_out = (const float*)d_in[5];
  const float* b_out = (const float*)d_in[6];

  __bf16* Winb  = (__bf16*)d_ws;
  __bf16* Wrecb = Winb + (size_t)HID * IN_DIM;
  __bf16* Woutb = Wrecb + (size_t)HID * HID;

  f2bf<<<512, 256, 0, stream>>>(W_in,  Winb,  HID * IN_DIM);
  f2bf<<<512, 256, 0, stream>>>(W_rec, Wrecb, HID * HID);
  f2bf<<<512, 256, 0, stream>>>(W_out, Woutb, OUT_DIM * HID);

  float* out    = (float*)d_out;
  float* interm = out + (size_t)OUT_DIM * BATCH;  // [33,1024,4096]

  const size_t lds_bytes =
      (size_t)(2 * BT * HSTRIDE + 2 * HID * WSTRIDE) * sizeof(__bf16); // ~289KB

  rnn_fused<<<BATCH / BT, NTHREADS, lds_bytes, stream>>>(
      x, Winb, b_in, Wrecb, b_rec, Woutb, b_out, out, interm);
}
